// MultiHeadSelfAttention_87351044866690
// MI455X (gfx1250) — matrix operand order repaired and checked by reading, not compiled, not run on hardware
//
#include <hip/hip_runtime.h>
#include <hip/hip_bf16.h>
#include <stdint.h>

// ---------------------------------------------------------------------------
// MultiHeadSelfAttention for MI455X (gfx1250, wave32, WMMA f16 -> f32 acc)
// B=4, S=2048, D=1024, H=16, HD=64
// Data movement: TDM tensor_load_to_lds (attention K/V tiles, double-buffered)
//                + global_load_async_to_lds_b128 (GEMM staging)
// ---------------------------------------------------------------------------

typedef _Float16 half_t;
typedef __attribute__((ext_vector_type(16))) _Float16 v16h;
typedef __attribute__((ext_vector_type(8)))  _Float16 v8h;
typedef __attribute__((ext_vector_type(4)))  _Float16 v4h;
typedef __attribute__((ext_vector_type(8)))  float    v8f;
typedef __attribute__((ext_vector_type(4)))  float    v4f;
typedef __attribute__((ext_vector_type(4)))  unsigned int u32x4;
typedef __attribute__((ext_vector_type(8)))  int      i32x8;
typedef __attribute__((ext_vector_type(4)))  int      i32x4;

#define B_  4
#define S_  2048
#define D_  1024
#define H_  16
#define HD_ 64

#if __has_builtin(__builtin_amdgcn_tensor_load_to_lds)
#define HAVE_TDM 1
#else
#define HAVE_TDM 0
#endif

__device__ __forceinline__ v16h cat16(v8h lo, v8h hi) {
  return __builtin_shufflevector(lo, hi, 0,1,2,3,4,5,6,7,8,9,10,11,12,13,14,15);
}

__device__ __forceinline__ v8f wmma_f16(v16h a, v16h b, v8f c) {
  return __builtin_amdgcn_wmma_f32_16x16x32_f16(false, a, false, b, (short)0, c,
                                                false, false);
}

// LDS byte offset of a __shared__ object (flat addr low 32 bits = LDS offset)
__device__ __forceinline__ uint32_t lds_off(const void* p) {
  return (uint32_t)(uintptr_t)p;
}

// Per-lane async copy global -> LDS, 16B, tracked with ASYNCcnt.
__device__ __forceinline__ void async_ld_b128(uint32_t ldsaddr, const void* g) {
  asm volatile("global_load_async_to_lds_b128 %0, %1, off"
               :: "v"(ldsaddr), "v"(g) : "memory");
}
__device__ __forceinline__ void wait_async0() {
#if __has_builtin(__builtin_amdgcn_s_wait_asynccnt)
  __builtin_amdgcn_s_wait_asynccnt(0);
#else
  asm volatile("s_wait_asynccnt 0x0" ::: "memory");
#endif
}
__device__ __forceinline__ void wait_async2() {
#if __has_builtin(__builtin_amdgcn_s_wait_asynccnt)
  __builtin_amdgcn_s_wait_asynccnt(2);
#else
  asm volatile("s_wait_asynccnt 0x2" ::: "memory");
#endif
}

#if HAVE_TDM
// 2-D TDM tile load: tile_d0 contiguous elements (f16) per row, tile_d1 rows,
// row stride stride0 (elements), LDS padding pad codes per ISA D# group1.
// 6-arg builtin: g0(u32x4), g1(i32x8), g2(i32x4), g3(i32x4), ext(i32x8), cpol.
__device__ __forceinline__ void tdm_load_2d(uint32_t ldsaddr, const void* gaddr,
                                            uint32_t tile_d0, uint32_t tile_d1,
                                            uint32_t tensor_d0, uint32_t tensor_d1,
                                            uint64_t stride0,
                                            uint32_t pad_int, uint32_t pad_amt) {
  uint64_t ga = (uint64_t)(uintptr_t)gaddr;
  u32x4 g0;
  g0[0] = 1u;                                           // count=1 user descriptor
  g0[1] = ldsaddr;                                      // lds_addr
  g0[2] = (uint32_t)(ga & 0xFFFFFFFFu);                 // global_addr lo
  g0[3] = (uint32_t)((ga >> 32) & 0x01FFFFFFu) | (2u << 30);  // addr hi | type=2
  i32x8 g1;
  g1[0] = (int)((1u << 16) |                            // data_size = 2B
                (1u << 20) |                            // pad_enable
                (pad_int << 22) | (pad_amt << 25));
  g1[1] = (int)((tensor_d0 & 0xFFFFu) << 16);           // tensor_dim0[15:0] @48
  g1[2] = (int)(((tensor_d0 >> 16) & 0xFFFFu) |         // tensor_dim0[31:16]
                ((tensor_d1 & 0xFFFFu) << 16));         // tensor_dim1[15:0]
  g1[3] = (int)(((tensor_d1 >> 16) & 0xFFFFu) |         // tensor_dim1[31:16]
                ((tile_d0 & 0xFFFFu) << 16));           // tile_dim0
  g1[4] = (int)(tile_d1 & 0xFFFFu);                     // tile_dim1 (tile_dim2=0)
  g1[5] = (int)(uint32_t)(stride0 & 0xFFFFFFFFu);       // tensor_dim0_stride lo
  g1[6] = (int)(uint32_t)((stride0 >> 32) & 0xFFFFu);   // stride hi
  g1[7] = 0;
  i32x4 z4 = {0, 0, 0, 0};
  i32x8 z8 = {0, 0, 0, 0, 0, 0, 0, 0};
  __builtin_amdgcn_tensor_load_to_lds(g0, g1, z4, z4, z8, 0);
}
#endif

// ---------------------------------------------------------------------------
// 1) hidden_states fp32 -> f16
// ---------------------------------------------------------------------------
__global__ void cvt_hs(const float* __restrict__ in, half_t* __restrict__ out, int n4) {
  int i = blockIdx.x * blockDim.x + threadIdx.x;
  if (i >= n4) return;
  v4f x = *(const v4f*)(in + (size_t)i * 4);
  v4h y;
  y[0] = (half_t)x[0]; y[1] = (half_t)x[1]; y[2] = (half_t)x[2]; y[3] = (half_t)x[3];
  *(v4h*)(out + (size_t)i * 4) = y;
}

// ---------------------------------------------------------------------------
// 2) W fp32 [K][N] -> Wt f16 [N][K] (LDS-tiled transpose)
// ---------------------------------------------------------------------------
__global__ void cvt_w_t(const float* __restrict__ w, half_t* __restrict__ wt) {
  __shared__ float tile[32][33];
  const int nb = blockIdx.x * 32, kb = blockIdx.y * 32;
  const int tx = threadIdx.x, ty = threadIdx.y;   // (32, 8)
#pragma unroll
  for (int j = 0; j < 32; j += 8)
    tile[ty + j][tx] = w[(size_t)(kb + ty + j) * D_ + nb + tx];
  __syncthreads();
#pragma unroll
  for (int j = 0; j < 32; j += 8)
    wt[(size_t)(nb + ty + j) * D_ + kb + tx] = (half_t)tile[tx][ty + j];
}

// ---------------------------------------------------------------------------
// 3) QKV GEMM: Y = X[8192,1024] @ W (+bias)*scale. Block 256 thr = 8 waves
//    (2x4); block tile 64x256; wave tile 32x64 (2x4 WMMA); BK=32; LDS double
//    buffered via global_load_async_to_lds_b128, k-loop unrolled x2 over the
//    two buffers (branch-free steady state, tail peeled).
// ---------------------------------------------------------------------------
__launch_bounds__(256)
__global__ void qkv_gemm(const half_t* __restrict__ X, const half_t* __restrict__ Wt,
                         const float* __restrict__ bias, half_t* __restrict__ out,
                         int vmode, float scale)
{
  constexpr int BM = 64, BN = 256, BK = 32, LDK = BK + 8;  // 80B row stride
  __shared__ half_t sA[2][BM][LDK];
  __shared__ half_t sB[2][BN][LDK];

  const int tid  = threadIdx.x;
  const int lane = tid & 31, wave = tid >> 5;
  const int wm = wave >> 2, wn = wave & 3;
  const int mblk = blockIdx.x * BM, nblk = blockIdx.y * BN;
  const int nlane = lane & 15, hl = lane >> 4;

  const int arow = tid >> 2, acol = (tid & 3) << 3;   // A: 64 rows x 32 halfs

  auto stage = [&](int buf, int k0) {
    async_ld_b128(lds_off(&sA[buf][arow][acol]),
                  X + (size_t)(mblk + arow) * D_ + k0 + acol);
#pragma unroll
    for (int i = 0; i < 4; ++i) {                      // B: 256 rows x 32 halfs
      int c = tid + i * 256;
      int r = c >> 2, col = (c & 3) << 3;
      async_ld_b128(lds_off(&sB[buf][r][col]),
                    Wt + (size_t)(nblk + r) * D_ + k0 + col);
    }
  };
  auto fragA = [&](int buf, int rbase) -> v16h {
    const half_t* p = &sA[buf][rbase + nlane][0];
    return cat16(*(const v8h*)(p + hl * 8), *(const v8h*)(p + 16 + hl * 8));
  };
  auto fragB = [&](int buf, int cbase) -> v16h {
    const half_t* p = &sB[buf][cbase + nlane][hl * 8];
    return cat16(*(const v8h*)p, *(const v8h*)(p + 16));
  };

  v8f acc[2][4] = {};
  auto compute = [&](int buf) {
    v16h a0 = fragA(buf, wm * 32), a1 = fragA(buf, wm * 32 + 16);
#pragma unroll
    for (int ni = 0; ni < 4; ++ni) {
      v16h b = fragB(buf, wn * 64 + ni * 16);
      acc[0][ni] = wmma_f16(a0, b, acc[0][ni]);
      acc[1][ni] = wmma_f16(a1, b, acc[1][ni]);
    }
  };

  stage(0, 0);
  wait_async0();
  __syncthreads();

  for (int k0 = 0; k0 < D_ - 2 * BK; k0 += 2 * BK) {
    stage(1, k0 + BK);        // prefetch into buf1 while computing buf0
    compute(0);
    wait_async0();
    __syncthreads();
    stage(0, k0 + 2 * BK);    // prefetch into buf0 while computing buf1
    compute(1);
    wait_async0();
    __syncthreads();
  }
  stage(1, D_ - BK);          // tail: last two K-steps
  compute(0);
  wait_async0();
  __syncthreads();
  compute(1);

#pragma unroll
  for (int mi = 0; mi < 2; ++mi)
#pragma unroll
    for (int ni = 0; ni < 4; ++ni)
#pragma unroll
      for (int r = 0; r < 8; ++r) {
        int gm = mblk + wm * 32 + mi * 16 + r + hl * 8;
        int gn = nblk + wn * 64 + ni * 16 + nlane;
        float v = (acc[mi][ni][r] + bias[gn]) * scale;
        int b = gm >> 11, s = gm & (S_ - 1);
        int h = gn >> 6,  d = gn & (HD_ - 1);
        if (vmode) out[(((size_t)b * H_ + h) * HD_ + d) * S_ + s] = (half_t)v;
        else       out[(((size_t)b * H_ + h) * S_ + s) * HD_ + d] = (half_t)v;
      }
}

// ---------------------------------------------------------------------------
// 4) Fused flash attention. Grid (S/128, B*H), block 256 = 8 waves.
//    K/Vt tiles double-buffered via TDM (wave0 issues, TENSORcnt pipelined),
//    tile loop unrolled x2 over the buffers; async-to-LDS fallback.
// ---------------------------------------------------------------------------
__launch_bounds__(256)
__global__ void flash_attn(const half_t* __restrict__ Q, const half_t* __restrict__ K,
                           const half_t* __restrict__ Vt, float* __restrict__ out)
{
  constexpr int KT = 32;
  constexpr int LK = HD_ + 8;   // 144B row stride (TDM pad: 32 DW + 4 DW)
  constexpr int LV = KT + 8;    // 80B row stride  (TDM pad: 16 DW + 4 DW)
  __shared__ half_t sK[2][KT][LK];
  __shared__ half_t sV[2][HD_][LV];
  __shared__ half_t sP[8][16][LV];

  const int tid  = threadIdx.x;
  const int lane = tid & 31, wave = tid >> 5;
  const int nlane = lane & 15, hl = lane >> 4;
  const int bh = blockIdx.y;
  const int qbase = blockIdx.x * 128 + wave * 16;

  const half_t* Qb = Q  + (size_t)bh * S_ * HD_;
  const half_t* Kb = K  + (size_t)bh * S_ * HD_;
  const half_t* Vb = Vt + (size_t)bh * HD_ * S_;

  // Q fragments (16 rows x d=64 as two 16x32 A-frags); Q pre-scaled by 1/8.
  v16h qf[2];
#pragma unroll
  for (int c = 0; c < 2; ++c) {
    const half_t* p = Qb + (size_t)(qbase + nlane) * HD_ + c * 32;
    qf[c] = cat16(*(const v8h*)(p + hl * 8), *(const v8h*)(p + 16 + hl * 8));
  }

  v8f o[4] = {};
  float m[8], l[8];
#pragma unroll
  for (int r = 0; r < 8; ++r) { m[r] = -3.0e38f; l[r] = 0.0f; }

  constexpr float LOG2E = 1.44269504088896340736f;
  constexpr int NTILES = S_ / KT;   // 64 (even)

#if HAVE_TDM
  auto issue_tile = [&](int buf, int kt) {   // wave-uniform; EXEC-independent
    tdm_load_2d(lds_off(&sK[buf][0][0]), Kb + (size_t)kt * HD_,
                /*tile*/ HD_, KT, /*tensor*/ HD_, S_, /*stride*/ HD_,
                /*pad_int 32DW*/ 4, /*pad 4DW*/ 3);
    tdm_load_2d(lds_off(&sV[buf][0][0]), Vb + kt,
                /*tile*/ KT, HD_, /*tensor*/ S_, HD_, /*stride*/ S_,
                /*pad_int 16DW*/ 3, /*pad 4DW*/ 3);
  };
  if (wave == 0) issue_tile(0, 0);
#else
  const int krow = tid >> 3, kcol = (tid & 7) << 3;   // K tile 32x64
  const int vrow = tid >> 2, vcol = (tid & 3) << 3;   // Vt tile 64x32
  auto issue_tile = [&](int buf, int kt) {
    async_ld_b128(lds_off(&sK[buf][krow][kcol]),
                  Kb + (size_t)(kt + krow) * HD_ + kcol);
    async_ld_b128(lds_off(&sV[buf][vrow][vcol]),
                  Vb + (size_t)vrow * S_ + kt + vcol);
  };
  issue_tile(0, 0);
#endif

  // One key tile: S = QK^T (4 WMMA), online softmax, O += PV (4 WMMA).
  auto process = [&](int cur) {
    v8f s0 = {}, s1 = {};
#pragma unroll
    for (int c = 0; c < 2; ++c) {
      const half_t* p0 = &sK[cur][nlane][c * 32 + hl * 8];
      const half_t* p1 = &sK[cur][16 + nlane][c * 32 + hl * 8];
      v16h b0 = cat16(*(const v8h*)p0, *(const v8h*)(p0 + 16));
      v16h b1 = cat16(*(const v8h*)p1, *(const v8h*)(p1 + 16));
      s0 = wmma_f16(qf[c], b0, s0);
      s1 = wmma_f16(qf[c], b1, s1);
    }
#pragma unroll
    for (int r = 0; r < 8; ++r) {
      float mx = fmaxf(s0[r], s1[r]);
#pragma unroll
      for (int off = 1; off < 16; off <<= 1) mx = fmaxf(mx, __shfl_xor(mx, off, 32));
      float mn = fmaxf(m[r], mx);
      float alpha = exp2f((m[r] - mn) * LOG2E);
      float p0 = exp2f((s0[r] - mn) * LOG2E);
      float p1 = exp2f((s1[r] - mn) * LOG2E);
      float sum = p0 + p1;
#pragma unroll
      for (int off = 1; off < 16; off <<= 1) sum += __shfl_xor(sum, off, 32);
      m[r] = mn;
      l[r] = l[r] * alpha + sum;
#pragma unroll
      for (int d = 0; d < 4; ++d) o[d][r] *= alpha;
      sP[wave][r + 8 * hl][nlane]      = (half_t)p0;
      sP[wave][r + 8 * hl][16 + nlane] = (half_t)p1;
    }
    __builtin_amdgcn_wave_barrier();  // wave-local LDS write->read ordering
    const half_t* pp = &sP[wave][nlane][0];
    v16h pf = cat16(*(const v8h*)(pp + hl * 8), *(const v8h*)(pp + 16 + hl * 8));
#pragma unroll
    for (int d = 0; d < 4; ++d) {
      const half_t* pv = &sV[cur][d * 16 + nlane][hl * 8];
      v16h bf = cat16(*(const v8h*)pv, *(const v8h*)(pv + 16));
      o[d] = wmma_f16(pf, bf, o[d]);
    }
  };

  for (int t = 0; t < NTILES; t += 2) {
    // ---- tile t (buffer 0); prefetch t+1 into buffer 1 ----
#if HAVE_TDM
    if (wave == 0) {
      issue_tile(1, (t + 1) * KT);                 // t+1 < NTILES always here
      __builtin_amdgcn_s_wait_tensorcnt((short)2); // tile t landed
    }
#else
    issue_tile(1, (t + 1) * KT);
    wait_async2();
#endif
    __syncthreads();
    process(0);
    __syncthreads();

    // ---- tile t+1 (buffer 1); prefetch t+2 into buffer 0 ----
#if HAVE_TDM
    if (wave == 0) {
      if (t + 2 < NTILES) {
        issue_tile(0, (t + 2) * KT);
        __builtin_amdgcn_s_wait_tensorcnt((short)2);
      } else {
        __builtin_amdgcn_s_wait_tensorcnt((short)0);
      }
    }
#else
    if (t + 2 < NTILES) { issue_tile(0, (t + 2) * KT); wait_async2(); }
    else                { wait_async0(); }
#endif
    __syncthreads();
    process(1);
    __syncthreads();
  }

  // ---- normalize and write out[b, s, h*64 + d] (fp32) ----
  const int b = bh >> 4, h = bh & (H_ - 1);
  float* ob = out + (size_t)b * S_ * D_ + h * HD_;
#pragma unroll
  for (int r = 0; r < 8; ++r) {
    float inv = 1.0f / l[r];
    int srow = qbase + r + 8 * hl;
#pragma unroll
    for (int d = 0; d < 4; ++d)
      ob[(size_t)srow * D_ + d * 16 + nlane] = o[d][r] * inv;
  }
}

// ---------------------------------------------------------------------------
// Host launcher
// ---------------------------------------------------------------------------
extern "C" void kernel_launch(void* const* d_in, const int* in_sizes, int n_in,
                              void* d_out, int out_size, void* d_ws, size_t ws_size,
                              hipStream_t stream) {
  (void)in_sizes; (void)n_in; (void)out_size; (void)ws_size;

  const float* hs = (const float*)d_in[0];
  const float* Wq = (const float*)d_in[1];
  const float* bq = (const float*)d_in[2];
  const float* Wk = (const float*)d_in[3];
  const float* bk = (const float*)d_in[4];
  const float* Wv = (const float*)d_in[5];
  const float* bv = (const float*)d_in[6];
  float* out = (float*)d_out;

  char* ws = (char*)d_ws;
  half_t* hs16 = (half_t*)ws;                               // B*S*D f16
  half_t* wtq  = (half_t*)(ws + (size_t)B_ * S_ * D_ * 2);  // 3x D*D f16
  half_t* wtk  = wtq + (size_t)D_ * D_;
  half_t* wtv  = wtk + (size_t)D_ * D_;
  half_t* q16  = wtv + (size_t)D_ * D_;                     // [b,h,s,d]
  half_t* k16  = q16 + (size_t)B_ * S_ * D_;                // [b,h,s,d]
  half_t* vt16 = k16 + (size_t)B_ * S_ * D_;                // [b,h,d,s]

  const int n4 = B_ * S_ * D_ / 4;
  cvt_hs<<<(n4 + 255) / 256, 256, 0, stream>>>(hs, hs16, n4);

  dim3 tg(D_ / 32, D_ / 32), tb(32, 8);
  cvt_w_t<<<tg, tb, 0, stream>>>(Wq, wtq);
  cvt_w_t<<<tg, tb, 0, stream>>>(Wk, wtk);
  cvt_w_t<<<tg, tb, 0, stream>>>(Wv, wtv);

  dim3 gg(B_ * S_ / 64, D_ / 256);   // (128, 4)
  qkv_gemm<<<gg, 256, 0, stream>>>(hs16, wtq, bq, q16, 0, 0.125f); // fold 1/sqrt(64)
  qkv_gemm<<<gg, 256, 0, stream>>>(hs16, wtk, bk, k16, 0, 1.0f);
  qkv_gemm<<<gg, 256, 0, stream>>>(hs16, wtv, bv, vt16, 1, 1.0f);

  dim3 ag(S_ / 128, B_ * H_);        // (16, 64)
  flash_attn<<<ag, 256, 0, stream>>>(q16, k16, vt16, out);
}
